// Attention_28802050687173
// MI455X (gfx1250) — compile-verified, matrix operand order checked by reading, not run on hardware
//
#include <hip/hip_runtime.h>
#include <hip/hip_bf16.h>

typedef __bf16 bf16;
typedef __attribute__((ext_vector_type(16))) __bf16 v16bf;
typedef __attribute__((ext_vector_type(8)))  __bf16 v8bf;
typedef __attribute__((ext_vector_type(4)))  __bf16 v4bf;
typedef __attribute__((ext_vector_type(8)))  float  v8f;
typedef __attribute__((ext_vector_type(4)))  float  v4f;
typedef __attribute__((ext_vector_type(4)))  unsigned int u32x4;
typedef __attribute__((ext_vector_type(8)))  int    i32x8;
typedef __attribute__((ext_vector_type(4)))  int    i32x4;

static constexpr int B  = 4;
static constexpr int C  = 64;
static constexpr int C2 = 128;
static constexpr int C4 = 256;
static constexpr int H  = 256;
static constexpr int W  = 256;
static constexpr int N  = H * W;          // 65536 pixels per image

// ---------------------------------------------------------------------------
// WMMA fragment loader (bf16, wave32): two ds_load_b128 + register concat.
// Tiles stored row-major, K-contiguous, rows 16B-aligned; B operands staged
// transposed ([col][K]) so the same loader serves both A and B.
// ---------------------------------------------------------------------------
__device__ __forceinline__ v16bf ld_frag(const bf16* p, int ld) {
  const int lane = threadIdx.x & 31;
  const int row  = lane & 15;
  const int kb   = (lane & 16) ? 8 : 0;
  const bf16* q  = p + row * ld + kb;
  v8bf lo = *(const v8bf*)(q);
  v8bf hi = *(const v8bf*)(q + 16);
  return __builtin_shufflevector(lo, hi, 0, 1, 2, 3, 4, 5, 6, 7,
                                 8, 9, 10, 11, 12, 13, 14, 15);
}

#define WMMA_BF16(A, Bm, Cacc) \
  __builtin_amdgcn_wmma_f32_16x16x32_bf16(false, (A), false, (Bm), (short)0, (Cacc), false, false)

// ---------------------------------------------------------------------------
// K0: zero the Gram accumulation buffer (graph replays must be deterministic)
// ---------------------------------------------------------------------------
__global__ void k0_zero(float* __restrict__ P, int n) {
  int i = blockIdx.x * 256 + threadIdx.x;
  if (i < n) P[i] = 0.0f;
}

// ---------------------------------------------------------------------------
// K1: t[b, o, n] = sum_c w_in[o, c] * x[b, c, n]   (1x1 conv as bf16 WMMA GEMM)
//     A-fragments held in registers across all pixel tiles; two independent
//     pixel tiles in flight per wave to fill WMMA hazard slots.
// ---------------------------------------------------------------------------
static constexpr int K1_PX = 128;
static constexpr int LDW   = 72;
static constexpr int LDXT  = 72;
static constexpr int LDO1  = 136;

__global__ __launch_bounds__(256) void k1_pointwise(const float* __restrict__ x,
                                                    const float* __restrict__ w_in,
                                                    bf16* __restrict__ t) {
  __shared__ __align__(16) bf16 sW[C4 * LDW];       // [o][c]   36 KB
  __shared__ __align__(16) bf16 sXT[K1_PX * LDXT];  // [px][c]  18 KB (transposed)
  __shared__ __align__(16) bf16 sOut[C4 * LDO1];    // [o][px]  68 KB

  const int tid   = threadIdx.x;
  const int chunk = blockIdx.x;
  const int b     = chunk >> 9;                  // 512 chunks per batch
  const int n0    = (chunk & 511) * K1_PX;

  for (int i = tid; i < C4 * C / 4; i += 256) {
    const int o = i >> 4, c4 = (i & 15) * 4;
    v4f wv = *(const v4f*)(w_in + o * C + c4);
    v4bf wb = { (bf16)wv[0], (bf16)wv[1], (bf16)wv[2], (bf16)wv[3] };
    *(v4bf*)&sW[o * LDW + c4] = wb;
  }
  for (int i = tid; i < C * K1_PX / 4; i += 256) {
    const int c = i >> 5, p4 = (i & 31) * 4;
    v4f xv = *(const v4f*)(x + ((size_t)b * C + c) * N + n0 + p4);
#pragma unroll
    for (int j = 0; j < 4; ++j) sXT[(p4 + j) * LDXT + c] = (bf16)xv[j];
  }
  __syncthreads();

  const int wave = tid >> 5;
  const int lane = tid & 31;
  // wave handles ot = wave*2, wave*2+1 ; 8 pixel tiles each
  for (int oo = 0; oo < 2; ++oo) {
    const int ot = wave * 2 + oo;
    const v16bf a0 = ld_frag(&sW[(ot * 16) * LDW + 0],  LDW);
    const v16bf a1 = ld_frag(&sW[(ot * 16) * LDW + 32], LDW);
    for (int pt = 0; pt < 8; pt += 2) {
      v16bf b00 = ld_frag(&sXT[((pt + 0) * 16) * LDXT + 0],  LDXT);
      v16bf b10 = ld_frag(&sXT[((pt + 1) * 16) * LDXT + 0],  LDXT);
      v16bf b01 = ld_frag(&sXT[((pt + 0) * 16) * LDXT + 32], LDXT);
      v16bf b11 = ld_frag(&sXT[((pt + 1) * 16) * LDXT + 32], LDXT);
      v8f acc0 = {}, acc1 = {};
      acc0 = WMMA_BF16(a0, b00, acc0);   // two independent chains interleave
      acc1 = WMMA_BF16(a0, b10, acc1);
      acc0 = WMMA_BF16(a1, b01, acc0);
      acc1 = WMMA_BF16(a1, b11, acc1);
#pragma unroll
      for (int vi = 0; vi < 8; ++vi) {
        const int o = ot * 16 + vi + ((lane >> 4) << 3);
        sOut[o * LDO1 + (pt + 0) * 16 + (lane & 15)] = (bf16)acc0[vi];
        sOut[o * LDO1 + (pt + 1) * 16 + (lane & 15)] = (bf16)acc1[vi];
      }
    }
  }
  __syncthreads();

  for (int i = tid; i < C4 * (K1_PX / 8); i += 256) {
    const int o = i >> 4, ck = i & 15;
    v8bf val = *(const v8bf*)&sOut[o * LDO1 + ck * 8];
    *(v8bf*)(t + ((size_t)b * C4 + o) * N + n0 + ck * 8) = val;
  }
}

// ---------------------------------------------------------------------------
// K2: qv = dwconv3x3(t[:, :128]) * t[:, 128:]  (SAME zero padding, per-channel)
// ---------------------------------------------------------------------------
__global__ __launch_bounds__(256) void k2_dwgate(const bf16* __restrict__ t,
                                                 const float* __restrict__ w_dw,
                                                 bf16* __restrict__ qv) {
  __shared__ float sWd[C2 * 9];
  __shared__ bf16  sT[3][W + 2];
  const int b   = blockIdx.x >> 8;
  const int r   = blockIdx.x & 255;
  const int col = threadIdx.x;

  for (int i = threadIdx.x; i < C2 * 9; i += 256) sWd[i] = w_dw[i];
  const size_t base = (size_t)b * C4 * N;

  for (int ch = 0; ch < C2; ++ch) {
    const bf16* tc = t + base + (size_t)ch * N;
    __syncthreads();
#pragma unroll
    for (int dr = 0; dr < 3; ++dr) {
      const int rr = r + dr - 1;
      sT[dr][col + 1] = (rr >= 0 && rr < H) ? tc[rr * W + col] : (bf16)0.0f;
    }
    if (col < 2) {
      const int edge = col * (W + 1);
#pragma unroll
      for (int dr = 0; dr < 3; ++dr) sT[dr][edge] = (bf16)0.0f;
    }
    __syncthreads();

    float acc = 0.0f;
#pragma unroll
    for (int dr = 0; dr < 3; ++dr)
#pragma unroll
      for (int dc = 0; dc < 3; ++dc)
        acc += sWd[ch * 9 + dr * 3 + dc] * (float)sT[dr][col + dc];

    const float gate = (float)t[base + (size_t)(C2 + ch) * N + r * W + col];
    qv[((size_t)b * C2 + ch) * N + r * W + col] = (bf16)(acc * gate);
  }
}

// ---------------------------------------------------------------------------
// K3: P[b] += U U^T over a K-chunk, U = qv[b] in [128 x N].
//     Staging via the Tensor Data Mover when available: one D# descriptor
//     moves the 128x256 bf16 tile Global->LDS, with TDM padding reproducing
//     the LDU=264 bank-swizzle (pad 4 DWORDs every 128 DWORDs).
// ---------------------------------------------------------------------------
static constexpr int K3_CHUNK = 2048;
static constexpr int K3_STAGE = 256;
static constexpr int LDU      = 264;   // 256 + 8 elements (16B pad per 512B row)

__global__ __launch_bounds__(256) void k3_gram(const bf16* __restrict__ qv,
                                               float* __restrict__ P) {
  __shared__ __align__(16) bf16 sU[C2 * LDU];   // ~66 KB

  const int b    = blockIdx.x >> 5;
  const int k0   = (blockIdx.x & 31) * K3_CHUNK;
  const int tid  = threadIdx.x;
  const int wave = tid >> 5;
  const int lane = tid & 31;

  v8f acc[8];
#pragma unroll
  for (int i = 0; i < 8; ++i) acc[i] = (v8f){0, 0, 0, 0, 0, 0, 0, 0};

  for (int stage = 0; stage < K3_CHUNK / K3_STAGE; ++stage) {
    __syncthreads();
#if __has_builtin(__builtin_amdgcn_tensor_load_to_lds) && __has_builtin(__builtin_amdgcn_s_wait_tensorcnt)
    if (wave == 0) {
      const bf16* gsrc = qv + (size_t)b * C2 * N + (size_t)(k0 + stage * K3_STAGE);
      const unsigned long long ga = (unsigned long long)(uintptr_t)gsrc;
      const unsigned ldsAddr = (unsigned)(uintptr_t)&sU[0];   // LDS byte offset
      // ---- D# group 0 (CDNA5 ISA) ----
      u32x4 g0;
      g0[0] = 1u;                                            // count=1, user mode
      g0[1] = ldsAddr;                                       // lds_addr
      g0[2] = (unsigned)(ga & 0xffffffffu);                  // global_addr[31:0]
      g0[3] = (unsigned)((ga >> 32) & 0x01ffffffu)           // global_addr[56:32]
              | (2u << 30);                                  // type=2 (image)
      // ---- D# group 1 (§8.4): 2D tile 256(elem) x 128(rows), bf16 ----
      i32x8 g1;
      g1[0] = (1 << 16)        // data_size = 2 bytes
            | (1 << 20)        // pad_enable
            | (6 << 22)        // pad_interval: 128 DWORDs (=512B = one row)
            | (3 << 25);       // pad_amount:   4 DWORDs (=16B -> LDU=264)
      g1[1] = (int)((N & 0xffff) << 16);                     // tensor_dim0[15:0]
      g1[2] = (int)(((N >> 16) & 0xffff) | ((C2 & 0xffff) << 16)); // dim0 hi | dim1 lo
      g1[3] = (int)((K3_STAGE & 0xffff) << 16);              // dim1 hi=0 | tile_dim0=256
      g1[4] = (int)C2;                                       // tile_dim1=128, tile_dim2=0
      g1[5] = (int)N;                                        // tensor_dim0_stride[31:0]
      g1[6] = 0;                                             // stride hi | dim1_stride lo
      g1[7] = 0;
      i32x4 z4 = {0, 0, 0, 0};
  #if defined(__clang_major__) && __clang_major__ >= 23
      i32x8 z8 = {0, 0, 0, 0, 0, 0, 0, 0};
      __builtin_amdgcn_tensor_load_to_lds(g0, g1, z4, z4, z8, 0);
  #else
      __builtin_amdgcn_tensor_load_to_lds(g0, g1, z4, z4, 0);
  #endif
      __builtin_amdgcn_s_wait_tensorcnt(0);
    }
#else
    for (int i = tid; i < C2 * (K3_STAGE / 8); i += 256) {
      const int ch = i >> 5, ck = i & 31;
      *(v8bf*)&sU[ch * LDU + ck * 8] =
          *(const v8bf*)(qv + ((size_t)b * C2 + ch) * N + k0 + stage * K3_STAGE + ck * 8);
    }
#endif
    if (stage + 1 < K3_CHUNK / K3_STAGE) {
      const bf16* nxt = &qv[((size_t)b * C2 + (tid >> 1)) * N + k0 + (stage + 1) * K3_STAGE];
      __builtin_prefetch(nxt, 0, 0);
    }
    __syncthreads();

    for (int ks = 0; ks < K3_STAGE / 32; ++ks) {
#pragma unroll
      for (int i = 0; i < 8; ++i) {
        const int tI = wave * 8 + i;          // 8x8 grid of 16x16 tiles
        const int ct = tI >> 3, dt = tI & 7;
        v16bf a  = ld_frag(&sU[(ct * 16) * LDU + ks * 32], LDU);
        v16bf bt = ld_frag(&sU[(dt * 16) * LDU + ks * 32], LDU);  // U^T operand
        acc[i] = WMMA_BF16(a, bt, acc[i]);
      }
    }
  }

#pragma unroll
  for (int i = 0; i < 8; ++i) {
    const int tI = wave * 8 + i;
    const int ct = tI >> 3, dt = tI & 7;
    const int colD = dt * 16 + (lane & 15);
#pragma unroll
    for (int vi = 0; vi < 8; ++vi) {
      const int rowC = ct * 16 + vi + ((lane >> 4) << 3);
      atomicAdd(&P[((size_t)b * C2 + rowC) * C2 + colD], acc[i][vi]);
    }
  }
}

// ---------------------------------------------------------------------------
// K4: norms -> normalized logits -> softmax -> M = w_out @ attn  (tiny, per b)
// ---------------------------------------------------------------------------
__global__ __launch_bounds__(64) void k4_softmax(const float* __restrict__ P,
                                                 const float* __restrict__ w_out,
                                                 const float* __restrict__ temperature,
                                                 bf16* __restrict__ Mout) {
  const int b = blockIdx.x;
  const int c = threadIdx.x;
  __shared__ float rq[64], rv[64];
  __shared__ float sAttn[64][64];

  const float* Pb = P + (size_t)b * C2 * C2;
  rq[c] = fmaxf(sqrtf(fmaxf(Pb[c * C2 + c], 0.0f)), 1e-12f);
  rv[c] = fmaxf(sqrtf(fmaxf(Pb[(64 + c) * C2 + 64 + c], 0.0f)), 1e-12f);
  __syncthreads();

  const float T = temperature[0];
  float srow[64];
  float mx = -3.0e38f;
  for (int d = 0; d < 64; ++d) {
    const float s = Pb[c * C2 + 64 + d] / (rq[c] * rv[d]) * T;
    srow[d] = s;
    mx = fmaxf(mx, s);
  }
  float sum = 0.0f;
  for (int d = 0; d < 64; ++d) { const float e = __expf(srow[d] - mx); srow[d] = e; sum += e; }
  const float inv = 1.0f / sum;
  for (int d = 0; d < 64; ++d) sAttn[c][d] = srow[d] * inv;
  __syncthreads();

  for (int d = 0; d < 64; ++d) {
    float m = 0.0f;
    for (int cc = 0; cc < 64; ++cc) m += w_out[c * 64 + cc] * sAttn[cc][d];
    Mout[((size_t)b * 64 + c) * 64 + d] = (bf16)m;
  }
}

// ---------------------------------------------------------------------------
// K5: out[b, o, n] = sum_d M[b, o, d] * v[b, d, n]   (WMMA GEMM, fp32 out)
//     A-fragments in registers; two pixel tiles in flight per wave.
// ---------------------------------------------------------------------------
static constexpr int K5_PX = 256;
static constexpr int LDM   = 72;
static constexpr int LDVT  = 72;
static constexpr int LDO5  = 260;

__global__ __launch_bounds__(256) void k5_out(const bf16* __restrict__ Mout,
                                              const bf16* __restrict__ qv,
                                              float* __restrict__ out) {
  __shared__ __align__(16) bf16  sM[C * LDM];        // [o][d]    9 KB
  __shared__ __align__(16) bf16  sVT[K5_PX * LDVT];  // [px][d]  36 KB (transposed)
  __shared__ __align__(16) float sOutF[C * LDO5];    // [o][px]  65 KB

  const int b    = blockIdx.x >> 8;
  const int n0   = (blockIdx.x & 255) * K5_PX;
  const int tid  = threadIdx.x;
  const int wave = tid >> 5;
  const int lane = tid & 31;

  for (int i = tid; i < C * C / 8; i += 256) {
    const int o = i >> 3, dk = (i & 7) * 8;
    *(v8bf*)&sM[o * LDM + dk] = *(const v8bf*)(Mout + (size_t)b * 4096 + o * 64 + dk);
  }
  for (int i = tid; i < C * K5_PX / 8; i += 256) {
    const int d = i >> 5, p8 = (i & 31) * 8;
    v8bf vv = *(const v8bf*)(qv + ((size_t)b * C2 + 64 + d) * N + n0 + p8);
#pragma unroll
    for (int j = 0; j < 8; ++j) sVT[(p8 + j) * LDVT + d] = vv[j];
  }
  __syncthreads();

  // wave -> (ot = wave/2, pixel tiles [(wave&1)*8, +8) of 16), 2 in flight
  const int ot  = wave >> 1;
  const int ptb = (wave & 1) * 8;
  const v16bf a0 = ld_frag(&sM[(ot * 16) * LDM + 0],  LDM);
  const v16bf a1 = ld_frag(&sM[(ot * 16) * LDM + 32], LDM);
  for (int pp = 0; pp < 8; pp += 2) {
    const int pt = ptb + pp;
    v16bf b00 = ld_frag(&sVT[((pt + 0) * 16) * LDVT + 0],  LDVT);
    v16bf b10 = ld_frag(&sVT[((pt + 1) * 16) * LDVT + 0],  LDVT);
    v16bf b01 = ld_frag(&sVT[((pt + 0) * 16) * LDVT + 32], LDVT);
    v16bf b11 = ld_frag(&sVT[((pt + 1) * 16) * LDVT + 32], LDVT);
    v8f acc0 = {}, acc1 = {};
    acc0 = WMMA_BF16(a0, b00, acc0);
    acc1 = WMMA_BF16(a0, b10, acc1);
    acc0 = WMMA_BF16(a1, b01, acc0);
    acc1 = WMMA_BF16(a1, b11, acc1);
#pragma unroll
    for (int vi = 0; vi < 8; ++vi) {
      const int o = ot * 16 + vi + ((lane >> 4) << 3);
      sOutF[o * LDO5 + (pt + 0) * 16 + (lane & 15)] = acc0[vi];
      sOutF[o * LDO5 + (pt + 1) * 16 + (lane & 15)] = acc1[vi];
    }
  }
  __syncthreads();

  for (int i = tid; i < C * (K5_PX / 4); i += 256) {
    const int o = i >> 6, ck = i & 63;
    v4f val = *(const v4f*)&sOutF[o * LDO5 + ck * 4];
    *(v4f*)(out + ((size_t)b * C + o) * N + n0 + ck * 4) = val;
  }
}

// ---------------------------------------------------------------------------
// Launch
// ---------------------------------------------------------------------------
extern "C" void kernel_launch(void* const* d_in, const int* in_sizes, int n_in,
                              void* d_out, int out_size, void* d_ws, size_t ws_size,
                              hipStream_t stream) {
  const float* x    = (const float*)d_in[0];
  const float* w_in = (const float*)d_in[1];
  const float* w_dw = (const float*)d_in[2];
  const float* w_out= (const float*)d_in[3];
  const float* temp = (const float*)d_in[4];
  float* out = (float*)d_out;

  char* ws = (char*)d_ws;
  size_t off = 0;
  bf16* t  = (bf16*)(ws + off);  off += (size_t)B * C4 * N * sizeof(bf16);   // 134 MB
  bf16* qv = (bf16*)(ws + off);  off += (size_t)B * C2 * N * sizeof(bf16);   //  67 MB
  float* P = (float*)(ws + off); off += (size_t)B * C2 * C2 * sizeof(float); // 256 KB
  bf16* M  = (bf16*)(ws + off);

  const int pElems = B * C2 * C2;
  k0_zero     <<<(pElems + 255) / 256, 256, 0, stream>>>(P, pElems);
  k1_pointwise<<<B * (N / K1_PX),      256, 0, stream>>>(x, w_in, t);
  k2_dwgate   <<<B * H,                256, 0, stream>>>(t, w_dw, qv);
  k3_gram     <<<B * 32,               256, 0, stream>>>(qv, P);
  k4_softmax  <<<B,                    64,  0, stream>>>(P, w_out, temp, M);
  k5_out      <<<B * (N / K5_PX),      256, 0, stream>>>(M, qv, out);
}
